// TemporalAttention_22514218566280
// MI455X (gfx1250) — matrix operand order repaired and checked by reading, not compiled, not run on hardware
//
#include <hip/hip_runtime.h>
#include <hip/hip_bf16.h>

#define HIDDEN 1024
#define FEAT   2048
#define BOT    512
#define BATCH  64
#define TLEN   512

typedef __attribute__((ext_vector_type(16))) __bf16 v16bf;
typedef __attribute__((ext_vector_type(8)))  __bf16 bf16x8;
typedef __attribute__((ext_vector_type(8)))  float  v8f;
typedef __attribute__((ext_vector_type(8)))  float  f32x8;

// branch-free tanh: saturates to +/-1, no NaN path
__device__ __forceinline__ float fast_tanh(float x) {
    float t = __expf(2.0f * x);
    return 1.0f - 2.0f / (t + 1.0f);
}

// ---------------- Phase 0: convert U (f32 row-major) -> Ubf (bf16 row-major) ----------------
// Row-major bf16 U is exactly the WMMA B-fragment layout: lane=col(n), K runs 8h..8h+7 and 16+8h..16+8h+7.
__global__ __launch_bounds__(256) void uconv_kernel(const float* __restrict__ U,
                                                    __bf16* __restrict__ Ubf) {
    const size_t i = ((size_t)blockIdx.x * 256 + threadIdx.x) * 8;  // 8 elems/thread
    f32x8 v = *(const f32x8*)(U + i);
    *(bf16x8*)(Ubf + i) = __builtin_convertvector(v, bf16x8);
}

// ---------------- Phase 1: Whb[b,n] = hidden[b,:] . W[n,:] + bias[n] ----------------
__global__ __launch_bounds__(256) void whb_kernel(const float* __restrict__ hidden,
                                                  const float* __restrict__ W,
                                                  const float* __restrict__ bias,
                                                  float* __restrict__ Whb) {
    const int b = blockIdx.y;
    const int n = blockIdx.x * 256 + threadIdx.x;   // grid.x = 2 -> n in [0,512)
    const float* h  = hidden + (size_t)b * HIDDEN;
    const float* wr = W + (size_t)n * HIDDEN;
    float acc = bias[n];
    #pragma unroll 4
    for (int k = 0; k < HIDDEN; k += 4) {
        acc += h[k] * wr[k] + h[k+1] * wr[k+1] + h[k+2] * wr[k+2] + h[k+3] * wr[k+3];
    }
    Whb[b * BOT + n] = acc;
}

// one 32-wide K step: convert A chunk, load 4 B fragments (one clause), 4 WMMAs
__device__ __forceinline__ void wmma_kstep(const __bf16* __restrict__ ub, int wave, int kb,
                                           f32x8 alo, f32x8 ahi, v8f* acc) {
    bf16x8 al = __builtin_convertvector(alo, bf16x8);
    bf16x8 ah = __builtin_convertvector(ahi, bf16x8);
    v16bf af = __builtin_shufflevector(al, ah, 0,1,2,3,4,5,6,7,8,9,10,11,12,13,14,15);

    v16bf bfr[4];
    #pragma unroll
    for (int j = 0; j < 4; ++j) {
        const __bf16* up = ub + (size_t)((wave + j * 8) * 16) * FEAT + kb;
        bf16x8 bl = *(const bf16x8*)(up);
        bf16x8 bh = *(const bf16x8*)(up + 16);
        bfr[j] = __builtin_shufflevector(bl, bh, 0,1,2,3,4,5,6,7,8,9,10,11,12,13,14,15);
    }
    #pragma unroll
    for (int j = 0; j < 4; ++j) {
        acc[j] = __builtin_amdgcn_wmma_f32_16x16x32_bf16(
            /*neg_a=*/false, af, /*neg_b=*/false, bfr[j],
            /*c_mod=*/(short)0, acc[j], /*reuse_a=*/false, /*reuse_b=*/false);
    }
}

// ---------------- Phase 2: WMMA GEMM + tanh + w-dot -> energies[b,t] ----------------
// block = 256 threads (8 waves). blockIdx.x = t-tile (32), blockIdx.y = b (64).
// wave w owns n-tiles {w, w+8, w+16, w+24}; K-loop over FEAT, 64/iteration, branch-free
// double-buffered A (two f32x8-pair buffers alternate; reloads land in loop-carried regs).
__global__ __launch_bounds__(256) void energy_kernel(const float* __restrict__ feats,
                                                     const __bf16* __restrict__ Ubf,
                                                     const float* __restrict__ Whb,
                                                     const float* __restrict__ wvec,
                                                     float* __restrict__ energies) {
    const int b     = blockIdx.y;
    const int mtile = blockIdx.x;
    const int tid   = threadIdx.x;
    const int wave  = tid >> 5;
    const int lane  = tid & 31;
    const int lhalf = lane >> 4;   // 0: lanes 0-15, 1: lanes 16-31
    const int l15   = lane & 15;

    // A-fragment row for this lane (feats row; bf16 16x32 A layout per ISA 7.12.2)
    const float* featRow = feats + ((size_t)b * TLEN + (size_t)mtile * 16 + l15) * FEAT;
    const float* ap = featRow + lhalf * 8;

    // B-fragment base for this lane: col = l15 within each n-tile, K in table order (8h+s, 16+8h+s)
    const __bf16* ub = Ubf + (size_t)l15 * FEAT + lhalf * 8;

    v8f acc[4] = {};   // 4 n-tiles x (16x16 f32 / 32 lanes = 8 VGPR)

    // preload two A chunks (kb=0 and kb=32)
    f32x8 a0lo = *(const f32x8*)(ap);
    f32x8 a0hi = *(const f32x8*)(ap + 16);
    f32x8 a1lo = *(const f32x8*)(ap + 32);
    f32x8 a1hi = *(const f32x8*)(ap + 48);

    int kb = 0;
    for (; kb < FEAT - 64; kb += 64) {
        wmma_kstep(ub, wave, kb, a0lo, a0hi, acc);        // consumes buffer 0
        a0lo = *(const f32x8*)(ap + kb + 64);             // refill buffer 0 (kb+64)
        a0hi = *(const f32x8*)(ap + kb + 80);
        wmma_kstep(ub, wave, kb + 32, a1lo, a1hi, acc);   // consumes buffer 1
        a1lo = *(const f32x8*)(ap + kb + 96);             // refill buffer 1 (kb+96)
        a1hi = *(const f32x8*)(ap + kb + 112);
    }
    // peeled tail: last 64 of K, no further A loads
    wmma_kstep(ub, wave, kb,      a0lo, a0hi, acc);
    wmma_kstep(ub, wave, kb + 32, a1lo, a1hi, acc);

    // ---- epilogue: e = tanh(acc + Whb); energy partial = e * w[n], reduce over n ----
    float epart[8] = {0.f, 0.f, 0.f, 0.f, 0.f, 0.f, 0.f, 0.f};
    #pragma unroll
    for (int j = 0; j < 4; ++j) {
        const int n    = (wave + j * 8) * 16 + l15;
        const float c  = Whb[b * BOT + n];
        const float ww = wvec[n];
        #pragma unroll
        for (int v = 0; v < 8; ++v) {
            float e = fast_tanh(acc[j][v] + c);
            epart[v] += e * ww;
        }
    }
    // butterfly over 16-lane half-groups (wave32): sums the 16 n-columns
    #pragma unroll
    for (int m = 1; m <= 8; m <<= 1) {
        #pragma unroll
        for (int v = 0; v < 8; ++v) epart[v] += __shfl_xor(epart[v], m, 32);
    }

    __shared__ float lds_e[8][16];
    if (l15 == 0) {
        #pragma unroll
        for (int v = 0; v < 8; ++v) lds_e[wave][lhalf * 8 + v] = epart[v];  // row = lhalf*8+v
    }
    __syncthreads();
    if (tid < 16) {
        float s = 0.f;
        #pragma unroll
        for (int w = 0; w < 8; ++w) s += lds_e[w][tid];
        energies[b * TLEN + mtile * 16 + tid] = s;
    }
}

// ---------------- Phase 3: softmax over T per batch ----------------
__global__ __launch_bounds__(512) void softmax_kernel(const float* __restrict__ energies,
                                                      float* __restrict__ weights) {
    __shared__ float red[512];
    const int b = blockIdx.x;
    const int t = threadIdx.x;
    const float e = energies[b * TLEN + t];
    red[t] = e;
    __syncthreads();
    for (int s = 256; s > 0; s >>= 1) {
        if (t < s) red[t] = fmaxf(red[t], red[t + s]);
        __syncthreads();
    }
    const float mx = red[0];
    __syncthreads();
    const float ex = __expf(e - mx);
    red[t] = ex;
    __syncthreads();
    for (int s = 256; s > 0; s >>= 1) {
        if (t < s) red[t] += red[t + s];
        __syncthreads();
    }
    weights[b * TLEN + t] = ex * (1.0f / red[0]);
}

// ---------------- Phase 4: attn_feats[b,f] = sum_t feats[b,t,f] * weights[b,t] ----------------
// Pure 256 MB stream over feats, each element read once -> non-temporal loads.
__global__ __launch_bounds__(256) void attnsum_kernel(const float* __restrict__ feats,
                                                      const float* __restrict__ weights,
                                                      float* __restrict__ out) {
    __shared__ float wl[TLEN];
    const int b = blockIdx.y;
    const int f = blockIdx.x * 256 + threadIdx.x;
    for (int t = threadIdx.x; t < TLEN; t += 256) wl[t] = weights[b * TLEN + t];
    __syncthreads();
    const float* fp = feats + (size_t)b * TLEN * FEAT + f;
    float acc = 0.f;
    #pragma unroll 4
    for (int t = 0; t < TLEN; ++t) acc += __builtin_nontemporal_load(fp + (size_t)t * FEAT) * wl[t];
    out[(size_t)b * FEAT + f] = acc;
}

extern "C" void kernel_launch(void* const* d_in, const int* in_sizes, int n_in,
                              void* d_out, int out_size, void* d_ws, size_t ws_size,
                              hipStream_t stream) {
    const float* hidden = (const float*)d_in[0];  // [64,1024]
    const float* feats  = (const float*)d_in[1];  // [64,512,2048]
    const float* W      = (const float*)d_in[2];  // [512,1024]
    const float* U      = (const float*)d_in[3];  // [512,2048]
    const float* bias   = (const float*)d_in[4];  // [512]
    const float* wvec   = (const float*)d_in[5];  // [1,512]

    float* out_attn = (float*)d_out;                       // [64,2048]
    float* out_wts  = (float*)d_out + BATCH * FEAT;        // [64,512,1]

    float*  Whb      = (float*)d_ws;                       // 32768 f32
    float*  energies = Whb + BATCH * BOT;                  // 32768 f32
    __bf16* Ubf      = (__bf16*)(energies + BATCH * TLEN); // 512*2048 bf16 (2 MB)

    uconv_kernel <<<dim3(BOT * FEAT / (256 * 8)), 256, 0, stream>>>(U, Ubf);
    whb_kernel   <<<dim3(2, BATCH),  256, 0, stream>>>(hidden, W, bias, Whb);
    energy_kernel<<<dim3(32, BATCH), 256, 0, stream>>>(feats, Ubf, Whb, wvec, energies);
    softmax_kernel<<<dim3(BATCH),    512, 0, stream>>>(energies, out_wts);
    attnsum_kernel<<<dim3(8, BATCH), 256, 0, stream>>>(feats, out_wts, out_attn);
}
